// CausalSelfAttention_6640019440342
// MI455X (gfx1250) — matrix operand order repaired and checked by reading, not compiled, not run on hardware
//
#include <hip/hip_runtime.h>

typedef __attribute__((ext_vector_type(16))) __bf16   v16bf;
typedef __attribute__((ext_vector_type(8)))  float    v8f;
typedef __attribute__((ext_vector_type(4)))  unsigned u32x4;
typedef __attribute__((ext_vector_type(4)))  int      v4i;

union Frag { v16bf v; u32x4 q[2]; };

// ---------------- gfx1250 async global->LDS path (guarded) ----------------
#if __has_builtin(__builtin_amdgcn_global_load_async_to_lds_b128) && \
    __has_builtin(__builtin_amdgcn_s_wait_asynccnt)
#define HAVE_ASYNC 1
typedef __attribute__((address_space(1))) v4i* gas_v4i_ptr;
typedef __attribute__((address_space(3))) v4i* las_v4i_ptr;
#else
#define HAVE_ASYNC 0
#endif

template <typename T>
__device__ __forceinline__ void copy16_to_lds(T* lds, const T* g) {
#if HAVE_ASYNC
    __builtin_amdgcn_global_load_async_to_lds_b128(
        (gas_v4i_ptr)const_cast<T*>(g),
        (las_v4i_ptr)lds, 0, 0);
#else
    *(u32x4*)lds = *(const u32x4*)g;
#endif
}
__device__ __forceinline__ void lds_copy_wait() {
#if HAVE_ASYNC
    __builtin_amdgcn_s_wait_asynccnt(0);
#endif
}

// ---------------- octonion sign table (compile-time) ----------------
struct OctSign { float s[8][8]; };

constexpr OctSign make_oct_sign() {
    int idx[8][8] = {}; int sgn[8][8] = {};
    idx[0][0] = 0; sgn[0][0] = 1;
    for (int n = 1; n < 8; n *= 2) {
        int I[8][8] = {}; int S[8][8] = {};
        for (int i = 0; i < n; ++i)
            for (int j = 0; j < n; ++j) {
                int cj = (j == 0) ? 1 : -1;
                I[i][j]     = idx[i][j];        S[i][j]     = sgn[i][j];
                I[i][n+j]   = n + idx[j][i];    S[i][n+j]   = sgn[j][i];
                I[n+i][j]   = n + idx[i][j];    S[n+i][j]   = sgn[i][j] * cj;
                I[n+i][n+j] = idx[j][i];        S[n+i][n+j] = -cj * sgn[j][i];
            }
        for (int i = 0; i < 2*n; ++i)
            for (int j = 0; j < 2*n; ++j) { idx[i][j] = I[i][j]; sgn[i][j] = S[i][j]; }
    }
    OctSign o = {};
    for (int i = 0; i < 8; ++i)
        for (int j = 0; j < 8; ++j)
            o.s[i][j] = (float)sgn[j][i ^ j];   // _SIGN[i][j] = _OSGN[j][i^j]
    return o;
}

__constant__ OctSign OCT = make_oct_sign();

// ---------------- helpers ----------------
__device__ __forceinline__ __bf16 f2bf(float f) {
    unsigned u = __builtin_bit_cast(unsigned, f);
    unsigned r = u + 0x7FFFu + ((u >> 16) & 1u);     // round-to-nearest-even
    return __builtin_bit_cast(__bf16, (unsigned short)(r >> 16));
}
__device__ __forceinline__ float bf2f(__bf16 h) {
    unsigned u = ((unsigned)__builtin_bit_cast(unsigned short, h)) << 16;
    return __builtin_bit_cast(float, u);
}
__device__ __forceinline__ v8f vzero8() {
    v8f z = {0.f,0.f,0.f,0.f,0.f,0.f,0.f,0.f};
    return z;
}
__device__ __forceinline__ v8f wmma_bf16(v16bf a, v16bf b, v8f c) {
    return __builtin_amdgcn_wmma_f32_16x16x32_bf16(false, a, false, b, (short)0, c, false, false);
}

// ---------------- conversion / layout kernels ----------------
__global__ void k_cvt_bf16(const float* __restrict__ in, __bf16* __restrict__ out, int n) {
    int i = blockIdx.x * blockDim.x + threadIdx.x;
    if (i < n) out[i] = f2bf(in[i]);
}

// in: (8, dim, dim) f32 row-major [m][p][q]  ->  out bf16 n-major [m][q][p]
__global__ void k_wtrans(const float* __restrict__ in, __bf16* __restrict__ out, int dim) {
    int i = blockIdx.x * blockDim.x + threadIdx.x;
    int per = dim * dim;
    if (i >= 8 * per) return;
    int m = i / per, rem = i - m * per;
    int q = rem / dim, p = rem - q * dim;
    out[i] = f2bf(in[m * per + p * dim + q]);
}

// RoPE: Q/K bf16 (T x C, head-major cols) -> per-head (H, T, D) bf16
__global__ void k_rope(const __bf16* __restrict__ Qin, const __bf16* __restrict__ Kin,
                       const float* __restrict__ cosT, const float* __restrict__ sinT,
                       __bf16* __restrict__ Qo, __bf16* __restrict__ Ko) {
    const int T = 2048, H = 32, D = 128, C = 4096, HF = 64;
    int i = blockIdx.x * blockDim.x + threadIdx.x;
    if (i >= T * H * HF) return;
    int k = i % HF; int rem = i / HF;
    int h = rem % H; int t = rem / H;
    float c = cosT[t * HF + k], s = sinT[t * HF + k];
    int src = t * C + h * D + 2 * k;
    size_t dst = (size_t)h * T * D + (size_t)t * D + 2 * k;
    float qe = bf2f(Qin[src]), qo = bf2f(Qin[src + 1]);
    Qo[dst]     = f2bf(qe * c - qo * s);
    Qo[dst + 1] = f2bf(qe * s + qo * c);
    float ke = bf2f(Kin[src]), ko = bf2f(Kin[src + 1]);
    Ko[dst]     = f2bf(ke * c - ko * s);
    Ko[dst + 1] = f2bf(ke * s + ko * c);
}

// V (T x C) -> Vt (H, D, T)
__global__ void k_vtrans(const __bf16* __restrict__ Vin, __bf16* __restrict__ Vt) {
    const int T = 2048, D = 128, C = 4096;
    int i = blockIdx.x * blockDim.x + threadIdx.x;
    if (i >= T * C) return;
    int t = i % T; int rem = i / T;
    int d = rem % D; int h = rem / D;
    Vt[i] = Vin[t * C + h * D + d];
}

// ---------------- structured oct GEMM ----------------
// OUT(T x 4096-block) = sum_j sign[i][j] * A[:, acb + j*KBS : ...] @ W[i^j]
// Sign is folded into the A tile so the weight tile is a pure (async) copy.
// MODE 0: proj (KBS=512) -> bf16 out.  MODE 1: proj -> f32 out.  MODE 2: mixer (KBS=128) -> bf16*beta.
template <int MODE>
__global__ __launch_bounds__(256)
void k_oct_gemm(const __bf16* __restrict__ X, const __bf16* __restrict__ Wt,
                __bf16* __restrict__ outB, float* __restrict__ outF,
                const float* __restrict__ beta) {
    constexpr int KBS      = (MODE == 2) ? 128 : 512;
    constexpr int Ktotal   = (MODE == 2) ? 1024 : 4096;
    constexpr int matElems = KBS * KBS;
    constexpr int lda = 4096, ldo = 4096;

    const int bx = blockIdx.x, by = blockIdx.y;
    const int m0 = by * 128;
    const int outColBase = bx * 128;
    int iIdx, q0, acb;
    if (MODE == 2) { iIdx = bx & 7; q0 = 0; acb = (bx >> 3) * 1024; }
    else           { iIdx = bx >> 2; q0 = (bx & 3) * 128; acb = 0; }

    const int tid  = threadIdx.x;
    const int wave = tid >> 5, lane = tid & 31;
    const int half = lane >> 4, lq = lane & 15;
    const int wm = wave >> 1, wn = wave & 1;   // 4 x 2 wave grid

    __shared__ __align__(16) __bf16 As[128][32];
    __shared__ __align__(16) __bf16 Bs[128][32];   // n-major

    v8f acc[2][4];
#pragma unroll
    for (int a = 0; a < 2; ++a)
#pragma unroll
        for (int b = 0; b < 4; ++b) acc[a][b] = vzero8();

    for (int k0 = 0; k0 < Ktotal; k0 += 32) {
        const int j = k0 / KBS;
        const __bf16* Wsrc = Wt + (size_t)(iIdx ^ j) * matElems;
        const bool neg = OCT.s[iIdx][j] < 0.f;
        const int p0 = k0 - j * KBS;
#pragma unroll
        for (int it = 0; it < 2; ++it) {
            int chunk = tid + it * 256;            // 512 chunks of 8 bf16
            int row = chunk >> 2, kc = (chunk & 3) * 8;
            // A tile: sync load + sign fold + ds store
            u32x4 av = *(const u32x4*)&X[(size_t)(m0 + row) * lda + acb + k0 + kc];
            if (neg) av ^= 0x80008000u;
            *(u32x4*)&As[row][kc] = av;
            // B tile: pure copy -> async global->LDS on gfx1250
            copy16_to_lds(&Bs[row][kc],
                          &Wsrc[(size_t)(q0 + row) * KBS + p0 + kc]);
            // prefetch next k-tile toward the WGP
            if (k0 + 32 < Ktotal) {
                const int nk = k0 + 32;
                const int nj = nk / KBS;
                __builtin_prefetch(&X[(size_t)(m0 + row) * lda + acb + nk + kc], 0, 0);
                __builtin_prefetch(&Wt[(size_t)(iIdx ^ nj) * matElems +
                                       (size_t)(q0 + row) * KBS + (nk - nj * KBS) + kc], 0, 0);
            }
        }
        lds_copy_wait();
        __syncthreads();

        Frag a[2], b[4];
#pragma unroll
        for (int mt = 0; mt < 2; ++mt) {
            int m = wm * 32 + mt * 16 + lq;
            a[mt].q[0] = *(const u32x4*)&As[m][half * 8];
            a[mt].q[1] = *(const u32x4*)&As[m][half * 8 + 16];
        }
#pragma unroll
        for (int nt = 0; nt < 4; ++nt) {
            int n = wn * 64 + nt * 16 + lq;
            b[nt].q[0] = *(const u32x4*)&Bs[n][half * 8];
            b[nt].q[1] = *(const u32x4*)&Bs[n][half * 8 + 16];
        }
#pragma unroll
        for (int mt = 0; mt < 2; ++mt)
#pragma unroll
            for (int nt = 0; nt < 4; ++nt)
                acc[mt][nt] = wmma_bf16(a[mt].v, b[nt].v, acc[mt][nt]);
        __syncthreads();
    }

#pragma unroll
    for (int mt = 0; mt < 2; ++mt)
#pragma unroll
        for (int nt = 0; nt < 4; ++nt) {
            int nloc = wn * 64 + nt * 16 + lq;
            float bv = (MODE == 2) ? beta[nloc] : 1.f;
#pragma unroll
            for (int r = 0; r < 8; ++r) {
                int row = m0 + wm * 32 + mt * 16 + r + 8 * half;
                int col = outColBase + nloc;
                float v = acc[mt][nt][r] * bv;
                if (MODE == 1) outF[(size_t)row * ldo + col] = v;
                else           outB[(size_t)row * ldo + col] = f2bf(v);
            }
        }
}

// ---------------- flash attention (per head, 64 q rows / block) ----------------
__global__ __launch_bounds__(128)
void k_attn(const __bf16* __restrict__ Qh, const __bf16* __restrict__ Kh,
            const __bf16* __restrict__ Vth, __bf16* __restrict__ Yattn) {
    const int T = 2048, D = 128, C = 4096;
    const int h = blockIdx.y, qb = blockIdx.x, q0 = qb * 64;
    const int tid = threadIdx.x;
    const int wave = tid >> 5, lane = tid & 31;
    const int half = lane >> 4, lq = lane & 15;

    __shared__ __align__(16) __bf16 Qs[64][128];
    __shared__ __align__(16) __bf16 Ks[64][128];
    __shared__ __align__(16) __bf16 Vs[128][64];   // n-major: [d][t]
    __shared__ __align__(16) __bf16 Ps[4][16][64];

    const __bf16* Qp = Qh + (size_t)h * T * D;
    const __bf16* Kp = Kh + (size_t)h * T * D;
    const __bf16* Vp = Vth + (size_t)h * D * T;

#pragma unroll
    for (int it = 0; it < 8; ++it) {
        int chunk = tid + it * 128;                // 1024 chunks of 8
        int r = chunk >> 4, c = (chunk & 15) * 8;
        copy16_to_lds(&Qs[r][c], &Qp[(size_t)(q0 + r) * D + c]);
    }

    v8f o[8];
#pragma unroll
    for (int i = 0; i < 8; ++i) o[i] = vzero8();
    float rmax[8], rsum[8];
#pragma unroll
    for (int r = 0; r < 8; ++r) { rmax[r] = -3.0e38f; rsum[r] = 0.f; }

    const float scale = 0.08838834764831845f;      // 1/sqrt(128)

    for (int kt = 0; kt <= qb; ++kt) {
        const int kbase = kt * 64;
        __syncthreads();
#pragma unroll
        for (int it = 0; it < 8; ++it) {
            int chunk = tid + it * 128;
            int r = chunk >> 4, c = (chunk & 15) * 8;
            copy16_to_lds(&Ks[r][c], &Kp[(size_t)(kbase + r) * D + c]);
        }
#pragma unroll
        for (int it = 0; it < 8; ++it) {
            int chunk = tid + it * 128;
            int r = chunk >> 3, c = (chunk & 7) * 8;
            copy16_to_lds(&Vs[r][c], &Vp[(size_t)r * T + kbase + c]);
        }
        lds_copy_wait();
        __syncthreads();

        // S = Q K^T  (16 x 64 per wave)
        v8f s[4];
#pragma unroll
        for (int i = 0; i < 4; ++i) s[i] = vzero8();
#pragma unroll
        for (int dk = 0; dk < 4; ++dk) {
            Frag a;
            int m = wave * 16 + lq;
            a.q[0] = *(const u32x4*)&Qs[m][dk * 32 + half * 8];
            a.q[1] = *(const u32x4*)&Qs[m][dk * 32 + half * 8 + 16];
#pragma unroll
            for (int nt = 0; nt < 4; ++nt) {
                Frag b;
                int n = nt * 16 + lq;
                b.q[0] = *(const u32x4*)&Ks[n][dk * 32 + half * 8];
                b.q[1] = *(const u32x4*)&Ks[n][dk * 32 + half * 8 + 16];
                s[nt] = wmma_bf16(a.v, b.v, s[nt]);
            }
        }

        // mask + online softmax
        float pm[4][8];
#pragma unroll
        for (int nt = 0; nt < 4; ++nt) {
            int key = kbase + nt * 16 + lq;
#pragma unroll
            for (int r = 0; r < 8; ++r) {
                int qrow = q0 + wave * 16 + r + 8 * half;
                float v = s[nt][r] * scale;
                pm[nt][r] = (key > qrow) ? -3.0e38f : v;
            }
        }
#pragma unroll
        for (int r = 0; r < 8; ++r) {
            float m4 = fmaxf(fmaxf(pm[0][r], pm[1][r]), fmaxf(pm[2][r], pm[3][r]));
#pragma unroll
            for (int off = 1; off < 16; off <<= 1)
                m4 = fmaxf(m4, __shfl_xor(m4, off, 32));
            float nm = fmaxf(rmax[r], m4);
            float corr = __expf(rmax[r] - nm);
            rmax[r] = nm;
            float ls = 0.f;
#pragma unroll
            for (int nt = 0; nt < 4; ++nt) {
                float p = __expf(pm[nt][r] - nm);
                pm[nt][r] = p;
                ls += p;
            }
#pragma unroll
            for (int off = 1; off < 16; off <<= 1)
                ls += __shfl_xor(ls, off, 32);
            rsum[r] = rsum[r] * corr + ls;
#pragma unroll
            for (int nt = 0; nt < 8; ++nt) o[nt][r] *= corr;
        }
#pragma unroll
        for (int nt = 0; nt < 4; ++nt)
#pragma unroll
            for (int r = 0; r < 8; ++r)
                Ps[wave][r + 8 * half][nt * 16 + lq] = f2bf(pm[nt][r]);
        __syncthreads();

        // O += P V   (16 x 128 per wave)
#pragma unroll
        for (int dk = 0; dk < 2; ++dk) {
            Frag a;
            int m = lq;
            a.q[0] = *(const u32x4*)&Ps[wave][m][dk * 32 + half * 8];
            a.q[1] = *(const u32x4*)&Ps[wave][m][dk * 32 + half * 8 + 16];
#pragma unroll
            for (int nt = 0; nt < 8; ++nt) {
                Frag b;
                int n = nt * 16 + lq;
                b.q[0] = *(const u32x4*)&Vs[n][dk * 32 + half * 8];
                b.q[1] = *(const u32x4*)&Vs[n][dk * 32 + half * 8 + 16];
                o[nt] = wmma_bf16(a.v, b.v, o[nt]);
            }
        }
    }

#pragma unroll
    for (int nt = 0; nt < 8; ++nt)
#pragma unroll
        for (int r = 0; r < 8; ++r) {
            int qrow = q0 + wave * 16 + r + 8 * half;
            int col = h * 128 + nt * 16 + lq;
            Yattn[(size_t)qrow * C + col] = f2bf(o[nt][r] / rsum[r]);
        }
}

// ---------------- launch ----------------
extern "C" void kernel_launch(void* const* d_in, const int* in_sizes, int n_in,
                              void* d_out, int out_size, void* d_ws, size_t ws_size,
                              hipStream_t stream) {
    (void)in_sizes; (void)n_in; (void)out_size; (void)ws_size;
    const float* x    = (const float*)d_in[0];
    const float* Wq   = (const float*)d_in[1];
    const float* Wk   = (const float*)d_in[2];
    const float* Wv   = (const float*)d_in[3];
    const float* Wo   = (const float*)d_in[4];
    const float* Wm   = (const float*)d_in[5];
    const float* beta = (const float*)d_in[6];
    const float* fc   = (const float*)d_in[7];
    const float* fs   = (const float*)d_in[8];
    float* out = (float*)d_out;

    const int T = 2048, C = 4096;
    const size_t MB = (size_t)1 << 20;
    char* ws = (char*)d_ws;
    __bf16* xb   = (__bf16*)(ws + 0 * MB);     // 16 MB
    __bf16* Wqt  = (__bf16*)(ws + 16 * MB);    // 4 MB each
    __bf16* Wkt  = (__bf16*)(ws + 20 * MB);
    __bf16* Wvt  = (__bf16*)(ws + 24 * MB);
    __bf16* Wot  = (__bf16*)(ws + 28 * MB);
    __bf16* Wmt  = (__bf16*)(ws + 32 * MB);    // 0.25 MB
    __bf16* Qr   = (__bf16*)(ws + 33 * MB);    // 16 MB each
    __bf16* Kr   = (__bf16*)(ws + 49 * MB);
    __bf16* Vr   = (__bf16*)(ws + 65 * MB);
    __bf16* Qrot = (__bf16*)(ws + 81 * MB);
    __bf16* Krot = (__bf16*)(ws + 97 * MB);
    __bf16* Vtr  = (__bf16*)(ws + 113 * MB);
    __bf16* Yatt = Qr;                          // reuse (dead after rope)
    __bf16* Ymix = Kr;                          // reuse

    const int nX = T * C;                       // 8388608
    k_cvt_bf16<<<nX / 256, 256, 0, stream>>>(x, xb, nX);
    k_wtrans<<<(8 * 512 * 512) / 256, 256, 0, stream>>>(Wq, Wqt, 512);
    k_wtrans<<<(8 * 512 * 512) / 256, 256, 0, stream>>>(Wk, Wkt, 512);
    k_wtrans<<<(8 * 512 * 512) / 256, 256, 0, stream>>>(Wv, Wvt, 512);
    k_wtrans<<<(8 * 512 * 512) / 256, 256, 0, stream>>>(Wo, Wot, 512);
    k_wtrans<<<(8 * 128 * 128) / 256, 256, 0, stream>>>(Wm, Wmt, 128);

    dim3 gG(32, 16), bG(256);
    k_oct_gemm<0><<<gG, bG, 0, stream>>>(xb, Wqt, Qr, nullptr, nullptr);
    k_oct_gemm<0><<<gG, bG, 0, stream>>>(xb, Wkt, Kr, nullptr, nullptr);
    k_oct_gemm<0><<<gG, bG, 0, stream>>>(xb, Wvt, Vr, nullptr, nullptr);

    k_rope<<<(T * 32 * 64) / 256, 256, 0, stream>>>(Qr, Kr, fc, fs, Qrot, Krot);
    k_vtrans<<<nX / 256, 256, 0, stream>>>(Vr, Vtr);

    dim3 gA(32, 32), bA(128);
    k_attn<<<gA, bA, 0, stream>>>(Qrot, Krot, Vtr, Yatt);

    k_oct_gemm<2><<<gG, bG, 0, stream>>>(Yatt, Wmt, Ymix, nullptr, beta);
    k_oct_gemm<1><<<gG, bG, 0, stream>>>(Ymix, Wot, nullptr, out, nullptr);
}
